// CONVClassifier_60790967108169
// MI455X (gfx1250) — matrix operand order repaired and checked by reading, not compiled, not run on hardware
//
#include <hip/hip_runtime.h>
#include <hip/hip_bf16.h>
#include <math.h>

// ---------------------------------------------------------------------------
// Types for WMMA fragments (gfx1250 wave32 WMMA: D = A(16x32 bf16) * B(32x16
// bf16) + C(16x16 f32), striped across the 32 lanes of one wave).
// NOTE: names chosen to avoid HIP's predefined ushort2/ushort4 vector types.
// ---------------------------------------------------------------------------
typedef __attribute__((ext_vector_type(8)))  __bf16         bf16x8;
typedef __attribute__((ext_vector_type(16))) __bf16         bf16x16;
typedef __attribute__((ext_vector_type(8)))  float          floatx8;
typedef __attribute__((ext_vector_type(8)))  unsigned short u16x8;
typedef __attribute__((ext_vector_type(4)))  unsigned short u16x4;
typedef __attribute__((ext_vector_type(2)))  unsigned short u16x2;
typedef __attribute__((ext_vector_type(2)))  float          f32x2;
typedef __attribute__((ext_vector_type(4)))  float          f32x4;

union FragBF { bf16x16 f; bf16x8 h[2]; };

__device__ __forceinline__ unsigned short f32_to_bf16(float f) {
  // round-to-nearest-even truncation to bf16 bits
  unsigned u = __float_as_uint(f);
  u += 0x7FFFu + ((u >> 16) & 1u);
  return (unsigned short)(u >> 16);
}

// ---------------------------------------------------------------------------
// Problem constants (CONV classifier):  V=50000 E=512 S=1024 NCLASS=2 B=64 L=256
// ---------------------------------------------------------------------------
#define EDIM   512
#define SDIM   1024
#define BATCH  64
#define SEQL   256
#define LROWS  255          // valid sliding-window rows per batch (L-1)

// ---------------------------------------------------------------------------
// Kernel 1: embedding gather + f32 -> bf16.   Gemb[b*L + l][0..511]
// grid = B*L blocks, 256 threads, 2 elements/thread (float2 -> u16x2).
// ---------------------------------------------------------------------------
__global__ void gather_embed_bf16(const int* __restrict__ idx,
                                  const float* __restrict__ emb,
                                  unsigned short* __restrict__ out) {
  const int row = blockIdx.x;             // 0 .. B*L-1
  const int t   = threadIdx.x;            // 0 .. 255
  const int tok = idx[row];
  const f32x2 v = ((const f32x2*)(emb + (long long)tok * EDIM))[t];
  u16x2 o;
  o.x = f32_to_bf16(v.x);
  o.y = f32_to_bf16(v.y);
  ((u16x2*)(out + (long long)row * EDIM))[t] = o;
}

// ---------------------------------------------------------------------------
// Kernel 2: Wc f32 -> bf16 (row-major copy).  grid = 1024 blocks x 256 thr,
// 4 elements/thread (float4 -> u16x4).  Wcb[s][0..1023]
// ---------------------------------------------------------------------------
__global__ void cvt_wc_bf16(const float* __restrict__ in,
                            unsigned short* __restrict__ out) {
  const long long i = (long long)blockIdx.x * 1024 + (long long)threadIdx.x * 4;
  const f32x4 v = *(const f32x4*)(in + i);
  u16x4 o;
  o.x = f32_to_bf16(v.x);
  o.y = f32_to_bf16(v.y);
  o.z = f32_to_bf16(v.z);
  o.w = f32_to_bf16(v.w);
  *(u16x4*)(out + i) = o;
}

// ---------------------------------------------------------------------------
// Kernel 3: fused pair-conv GEMM + max-pool + sigmoid.
//   conv[b,l,s] = sum_k A[l,k] * Wc[s,k] ,  A[l,k] = Gemb[b*256+l+(k>=512)][k&511]
//   sent[b,s]   = sigmoid( max_l conv[b,l,s] + bc[s] )
// grid  = (8 n-tiles of 128 cols, 64 batches), block = 256 (8 wave32s).
// Each wave owns 16 columns; M looped in 4 supertiles of 64 rows; K chunked
// by 128 through LDS.  16 v_wmma_f32_16x16x32_bf16 per (wave, chunk).
// __launch_bounds__(256, 2): cap VGPRs <= 256 (no s_set_vgpr_msb churn) and
// run 2 workgroups/WGP (104 KB of 320 KB LDS) for latency hiding.
// ---------------------------------------------------------------------------
#define KC   128
#define LDA  136           // 128 + 8-half pad (272B row stride -> conflict-free b128)
#define LDB  136

__global__ void __launch_bounds__(256, 2)
conv_maxpool_wmma(const unsigned short* __restrict__ Gemb,  // [B*256][512] bf16 bits
                  const unsigned short* __restrict__ Wcb,   // [1024][1024] bf16 bits
                  const float* __restrict__ bc,             // [1024]
                  float* __restrict__ sent) {               // [B][1024]
  __shared__ __align__(16) unsigned short sA[64][LDA];      // 17.4 KB
  __shared__ __align__(16) unsigned short sB[128][LDB];     // 34.8 KB

  const int tid   = threadIdx.x;
  const int b     = blockIdx.y;
  const int nbase = blockIdx.x * 128;
  const int wave  = tid >> 5;
  const int lane  = tid & 31;
  const int lr    = lane & 15;     // row/col within 16
  const int hi16  = lane >> 4;     // 0: K-low half, 1: K-high half

  // cooperative loader mapping: 16 rows x (16 x u16x8) per pass
  const int ldr = tid >> 4;        // 0..15
  const int ldc = (tid & 15) * 8;  // 0,8,..,120 (halfs)

  float rowmax = -3.0e38f;

  for (int msup = 0; msup < 4; ++msup) {
    floatx8 acc[4] = {};           // 4 m-tiles x 16x16 f32 accumulators

    for (int kc = 0; kc < 2 * EDIM; kc += KC) {
      const int hi  = kc >> 9;     // second token of the pair for k >= 512
      const int kcl = kc & (EDIM - 1);

      // ---- load A chunk: 64 rows x 128 halfs (row 255 clamped to 254) ----
      #pragma unroll
      for (int p = 0; p < 4; ++p) {
        const int r = p * 16 + ldr;
        int l = msup * 64 + r;
        if (l > LROWS - 1) l = LROWS - 1;
        const long long srow = (long long)b * SEQL + l + hi;
        *(u16x8*)&sA[r][ldc] =
            *(const u16x8*)(Gemb + srow * EDIM + kcl + ldc);
      }
      // ---- load B chunk: Wc rows nbase..nbase+127, cols kc..kc+127 ----
      #pragma unroll
      for (int p = 0; p < 8; ++p) {
        const int r = p * 16 + ldr;
        *(u16x8*)&sB[r][ldc] =
            *(const u16x8*)(Wcb + (long long)(nbase + r) * (2 * EDIM) + kc + ldc);
      }
      __syncthreads();

      // B-fragments for this wave's 16 columns (reused across 4 m-tiles).
      // B 32x16 bf16 layout: lanes 0-15 hold K=k0..k0+7 and k0+16..k0+23 of
      // column lr; lanes 16-31 hold K=k0+8..k0+15 and k0+24..k0+31.
      FragBF bf[4];
      #pragma unroll
      for (int ks = 0; ks < 4; ++ks) {
        const unsigned short* p = &sB[wave * 16 + lr][ks * 32 + hi16 * 8];
        bf[ks].h[0] = *(const bf16x8*)(p);
        bf[ks].h[1] = *(const bf16x8*)(p + 16);
      }

      // A-fragments, software-pipelined: issue loads for fragment i+1 before
      // the WMMA consuming fragment i so one LDS load stays in flight.
      // A 16x32 bf16 layout: lane(0-15) row=lr holds K=k0..k0+7 and
      // K=k0+16..k0+23; lane(16-31) holds K=k0+8..k0+15 and k0+24..k0+31.
      FragBF af;
      {
        const unsigned short* p = &sA[lr][hi16 * 8];
        af.h[0] = *(const bf16x8*)(p);
        af.h[1] = *(const bf16x8*)(p + 16);
      }
      #pragma unroll
      for (int it = 0; it < 16; ++it) {
        const int mt = it >> 2;
        const int ks = it & 3;
        FragBF afn;
        if (it < 15) {
          const int mt2 = (it + 1) >> 2;
          const int ks2 = (it + 1) & 3;
          const unsigned short* p = &sA[mt2 * 16 + lr][ks2 * 32 + hi16 * 8];
          afn.h[0] = *(const bf16x8*)(p);
          afn.h[1] = *(const bf16x8*)(p + 16);
        }
        acc[mt] = __builtin_amdgcn_wmma_f32_16x16x32_bf16(
            false, af.f, false, bf[ks].f, (short)0, acc[mt], false, false);
        if (it < 15) af = afn;
      }
      __syncthreads();
    }

    // fold this supertile's 64 rows into the running per-column max
    #pragma unroll
    for (int mt = 0; mt < 4; ++mt)
      #pragma unroll
      for (int j = 0; j < 8; ++j)
        rowmax = fmaxf(rowmax, acc[mt][j]);
  }

  // C/D layout: lane n (0-15) holds rows M=VGPRj, lane n+16 holds M=VGPRj+8
  // for the SAME column n -> combine halves across the wave (wave32 shuffle).
  rowmax = fmaxf(rowmax, __shfl_xor(rowmax, 16, 32));

  if (lane < 16) {
    const int n = nbase + wave * 16 + lr;
    const float x = rowmax + bc[n];
    sent[(long long)b * SDIM + n] = 1.0f / (1.0f + __expf(-x));
  }
}

// ---------------------------------------------------------------------------
// Kernel 4: head MLP 1024 -> 50 -> 2 + log_softmax.  One block per batch row.
// ---------------------------------------------------------------------------
__global__ void head_kernel(const float* __restrict__ sent,
                            const float* __restrict__ W1, const float* __restrict__ b1,
                            const float* __restrict__ W2, const float* __restrict__ b2,
                            float* __restrict__ out) {
  __shared__ float h[50];
  __shared__ float lg[2];
  const int b = blockIdx.x;
  const int t = threadIdx.x;

  if (t < 50) {
    float a = b1[t];
    const float* s = sent + (long long)b * SDIM;
    const float* w = W1 + (long long)t * SDIM;
    for (int i = 0; i < SDIM; ++i) a = fmaf(s[i], w[i], a);
    h[t] = a;
  }
  __syncthreads();
  if (t < 2) {
    float a = b2[t];
    for (int j = 0; j < 50; ++j) a = fmaf(h[j], W2[t * 50 + j], a);
    lg[t] = a;
  }
  __syncthreads();
  if (t == 0) {
    const float m   = fmaxf(lg[0], lg[1]);
    const float lse = m + logf(expf(lg[0] - m) + expf(lg[1] - m));
    out[b * 2 + 0] = lg[0] - lse;
    out[b * 2 + 1] = lg[1] - lse;
  }
}

// ---------------------------------------------------------------------------
// Launch
// ---------------------------------------------------------------------------
extern "C" void kernel_launch(void* const* d_in, const int* in_sizes, int n_in,
                              void* d_out, int out_size, void* d_ws, size_t ws_size,
                              hipStream_t stream) {
  const int*   inputs = (const int*)  d_in[0];   // [64,256]
  const float* emb    = (const float*)d_in[1];   // [50000,512]
  const float* Wc     = (const float*)d_in[2];   // [1024,1024]
  const float* bc     = (const float*)d_in[3];   // [1024]
  const float* W1     = (const float*)d_in[4];   // [50,1024]
  const float* b1     = (const float*)d_in[5];   // [50]
  const float* W2     = (const float*)d_in[6];   // [2,50]
  const float* b2     = (const float*)d_in[7];   // [2]
  float* out = (float*)d_out;                    // [64,2]

  char* ws = (char*)d_ws;
  const long long gembBytes = (long long)BATCH * SEQL * EDIM * 2;  // 16.8 MB
  const long long wcbBytes  = (long long)SDIM * 2 * EDIM * 2;      //  2.1 MB
  unsigned short* Gemb = (unsigned short*)ws;
  unsigned short* Wcb  = (unsigned short*)(ws + gembBytes);
  float*          sent = (float*)(ws + gembBytes + wcbBytes);      // 256 KB

  gather_embed_bf16<<<dim3(BATCH * SEQL), dim3(256), 0, stream>>>(inputs, emb, Gemb);
  cvt_wc_bf16      <<<dim3(SDIM),         dim3(256), 0, stream>>>(Wc, Wcb);
  conv_maxpool_wmma<<<dim3(8, BATCH),     dim3(256), 0, stream>>>(Gemb, Wcb, bc, sent);
  head_kernel      <<<dim3(BATCH),        dim3(64),  0, stream>>>(sent, W1, b1, W2, b2, out);
}
